// SequenceModel_44839458570573
// MI455X (gfx1250) — matrix operand order repaired and checked by reading, not compiled, not run on hardware
//
#include <hip/hip_runtime.h>
#include <hip/hip_bf16.h>
#include <math.h>
#include <stdint.h>

// ---------------------------------------------------------------------------
// MI455X (gfx1250) implementation notes:
//  * All GEMMs use v_wmma_f32_16x16x32_bf16 (wave32, 16x16 C tile, K=32).
//  * Weights are pre-transposed to [N,K] bf16 so B fragments are contiguous
//    32-byte runs -> global_load_b128.
//  * The 16-row A tile shared by all 8 waves of a block is staged into LDS
//    with gfx1250 async copies (global_load_async_to_lds_b64 + ASYNCcnt).
//  * Each wave register-blocks a 16x64 strip: 4 independent WMMA accumulators
//    (A-fragment reuse x4, WMMA latency overlap).
//  * Attention keeps the 16x1024 score strip per (b,h,qtile) entirely in LDS
//    (f32 scores + bf16 probabilities) -> no S^2 traffic to HBM.
//  * 192MB L2 swallows the whole working set (weights ~13MB bf16, acts <100MB).
// ---------------------------------------------------------------------------

typedef __bf16 bf16_t;
typedef __attribute__((ext_vector_type(16))) __bf16 v16bf;
typedef __attribute__((ext_vector_type(8)))  float  v8f;

#define DN    512
#define SN    1024
#define BN    16
#define HN    8
#define HDN   64
#define LN_   2
#define DFFN  2048
#define NCATN 1000
#define NCONTN 2048
#define MTOK  (BN * SN)      // 16384 tokens
#define MTGT  (BN * 512)     // 8192 target tokens

// ---------------- WMMA helpers (CDNA5 16x16x32 bf16) -----------------------

__device__ __forceinline__ v8f wmma_bf16(v16bf a, v16bf b, v8f c) {
  // 8 args: (neg_a, A, neg_b, B, c_mod, C, reuse_a, reuse_b)
  return __builtin_amdgcn_wmma_f32_16x16x32_bf16(false, a, false, b,
                                                 (short)0, c, false, false);
}

// A fragment: 16x32 bf16 tile, row-major source with `stride` elems per row.
__device__ __forceinline__ v16bf load_a_frag(const bf16_t* base, int stride,
                                             int lane) {
  const int half = lane >> 4;
  const int m    = lane & 15;
  const bf16_t* p = base + (size_t)m * stride;
  v16bf a;
#pragma unroll
  for (int j = 0; j < 8; ++j) {
    const int kk = half * 16 + 2 * j;
    a[2 * j]     = p[kk];
    a[2 * j + 1] = p[kk + 1];
  }
  return a;
}

// B fragment from a [K x N] row-major chunk: element (kk, n) = base[kk*stride+n]
__device__ __forceinline__ v16bf load_b_frag_kn(const bf16_t* base, int stride,
                                                int lane) {
  const int half = lane >> 4;
  const int n    = lane & 15;
  v16bf b;
#pragma unroll
  for (int j = 0; j < 8; ++j) {
    const int kk = half * 16 + 2 * j;
    b[2 * j]     = base[(size_t)kk * stride + n];
    b[2 * j + 1] = base[(size_t)(kk + 1) * stride + n];
  }
  return b;
}

// B fragment from an [N x K] row-major chunk: element (kk, n) = base[n*stride+kk]
// -> per lane the 16 elements are contiguous: lowers to 2x global_load_b128.
__device__ __forceinline__ v16bf load_b_frag_nk(const bf16_t* base, int stride,
                                                int lane) {
  const int half = lane >> 4;
  const int n    = lane & 15;
  const bf16_t* p = base + (size_t)n * stride + half * 16;
  v16bf b;
#pragma unroll
  for (int j = 0; j < 8; ++j) {
    b[2 * j]     = p[2 * j];
    b[2 * j + 1] = p[2 * j + 1];
  }
  return b;
}

// gfx1250 async copy: each lane moves 8 bytes global->LDS; one instruction per
// wave, tracked with ASYNCcnt.
__device__ __forceinline__ void async_g2lds_b64(uint32_t lds_byte_addr,
                                                const void* gaddr) {
  asm volatile("global_load_async_to_lds_b64 %0, %1, off"
               :: "v"(lds_byte_addr), "v"(gaddr)
               : "memory");
}
__device__ __forceinline__ void wait_async0() {
  asm volatile("s_wait_asynccnt 0x0" ::: "memory");
}

// ---------------- generic WMMA GEMM ----------------------------------------
// C[M,N] = act(A[M,K](bf16) @ Wt^T + bias), Wt is [N,K] row-major bf16.
// Block: 256 threads (8 waves) -> 16 x 512 output strip; each wave owns a
// 16x64 strip (4 WMMA accumulators).  A tile (16x64) staged in LDS via async.
template <int RELU>
__global__ __launch_bounds__(256) void k_gemm(
    const bf16_t* __restrict__ A, int lda,
    const bf16_t* __restrict__ Wt, int ldw,
    const float* __restrict__ bias,
    float* __restrict__ Cf, bf16_t* __restrict__ Cb, int ldc,
    int M, int N, int K) {
  __shared__ bf16_t sA[16 * 64];  // one K-stage of A (2 KB)
  const int tid  = threadIdx.x;
  const int lane = tid & 31;
  const int wave = tid >> 5;
  const int m0   = blockIdx.y * 16;
  const int nw0  = blockIdx.x * 512 + wave * 64;
  const int half = lane >> 4;
  const int nr   = lane & 15;

  // async staging: thread t copies 8B -> sA row (t>>4), elems (t&15)*4..+3
  const uint32_t sA_base = (uint32_t)(uintptr_t)(const void*)sA;
  const int arow = tid >> 4;
  const int acol = (tid & 15) * 4;
  const uint32_t lds_dst = sA_base + (uint32_t)tid * 8;

  v8f acc[4] = {};
  for (int k0 = 0; k0 < K; k0 += 64) {
    if (k0) __syncthreads();  // all waves done reading previous stage
    async_g2lds_b64(lds_dst, A + (size_t)(m0 + arow) * lda + k0 + acol);
    wait_async0();
    __syncthreads();
#pragma unroll
    for (int kc = 0; kc < 64; kc += 32) {
      const v16bf a = load_a_frag(sA + kc, 64, lane);  // ds_load_b128 x2
#pragma unroll
      for (int s = 0; s < 4; ++s) {
        const int n   = nw0 + s * 16 + nr;
        const int ncl = (n < N) ? n : (N - 1);  // clamp; stores are guarded
        const bf16_t* wp = Wt + (size_t)ncl * ldw + k0 + kc + half * 16;
        v16bf b;
#pragma unroll
        for (int j = 0; j < 16; ++j) b[j] = wp[j];  // 2x global_load_b128
        acc[s] = wmma_bf16(a, b, acc[s]);
      }
    }
  }
  // C layout: VGPR r holds row m = r + 8*half, col = n0 + (lane&15)
#pragma unroll
  for (int s = 0; s < 4; ++s) {
    const int n = nw0 + s * 16 + nr;
    if (n < N) {
      const float bv = bias ? bias[n] : 0.0f;
#pragma unroll
      for (int r = 0; r < 8; ++r) {
        const int m = m0 + r + 8 * half;
        float v = acc[s][r] + bv;
        if (RELU) v = fmaxf(v, 0.0f);
        if (Cf) Cf[(size_t)m * ldc + n] = v;
        if (Cb) Cb[(size_t)m * ldc + n] = (bf16_t)v;
      }
    }
  }
}

// ---------------- embedding ------------------------------------------------
__global__ __launch_bounds__(256) void k_embed(
    const int* __restrict__ ccat, const float* __restrict__ ctime,
    const float* __restrict__ cf0, const int* __restrict__ tcat,
    const float* __restrict__ ttime, const float* __restrict__ tf0,
    const float* __restrict__ cat_table, const float* __restrict__ pad_emb,
    float* __restrict__ xf, bf16_t* __restrict__ xb,
    float* __restrict__ biasv) {
  const int t = blockIdx.x;          // 0..16383
  const int b = t >> 10;
  const int pos = t & 1023;
  int cat; float tm, f0;
  if (pos < 512) {
    const int i = b * 512 + pos;
    cat = ccat[i]; tm = ctime[i]; f0 = cf0[i];
  } else {
    const int i = b * 512 + (pos - 512);
    cat = tcat[i]; tm = ttime[i]; f0 = tf0[i];
  }
  const bool pad = cat < 0;
  const int catc = pad ? 0 : cat;
  const size_t base = (size_t)t * DN;
  for (int d = threadIdx.x; d < DN; d += 256) {
    float val;
    if (d < 256) {
      val = cat_table[(size_t)catc * 256 + d];
    } else if (d < 384) {
      const int i = d - 256;
      const float ang = tm * powf(1000.0f, -(float)i * (1.0f / 128.0f));
      val = (i & 1) ? cosf(ang) : sinf(ang);
    } else {
      const int i = d - 384;
      const float ang = f0 * powf(1000.0f, -(float)i * (1.0f / 128.0f));
      val = (i & 1) ? cosf(ang) : sinf(ang);
    }
    if (pad) val = pad_emb[d];
    xf[base + d] = val;
    xb[base + d] = (bf16_t)val;
  }
  if (threadIdx.x == 0) biasv[t] = pad ? -1.0e9f : 0.0f;
}

// ---------------- f32 -> bf16 conversion (plain / transposing) -------------
__global__ void k_cvt(const float* __restrict__ s, bf16_t* __restrict__ d,
                      int n) {
  const int i = blockIdx.x * blockDim.x + threadIdx.x;
  if (i < n) d[i] = (bf16_t)s[i];
}

// s: [K,N] f32 row-major  ->  d: [N,K] bf16 row-major
__global__ void k_cvt_t(const float* __restrict__ s, bf16_t* __restrict__ d,
                        int K, int N) {
  const int j = blockIdx.x * blockDim.x + threadIdx.x;
  if (j < K * N) {
    const int n = j / K;
    const int k = j - n * K;
    d[j] = (bf16_t)s[(size_t)k * N + n];
  }
}

// ---------------- fused LDS attention --------------------------------------
// One workgroup (8 waves) per (qtile16, head, batch).  LDS: 16x1024 f32 scores
// + 16x1024 bf16 probs + reductions + split-K partials (~101 KB < 320 KB/WGP).
__global__ __launch_bounds__(256) void k_attn(
    const bf16_t* __restrict__ qb, const bf16_t* __restrict__ kb,
    const bf16_t* __restrict__ vb, const float* __restrict__ biasv,
    bf16_t* __restrict__ ob) {
  extern __shared__ char smem[];
  float*  sS   = (float*)smem;                                  // 16*1024 f32
  bf16_t* sP   = (bf16_t*)(smem + 16 * 1024 * 4);               // 16*1024 bf16
  float*  sRed = (float*)(smem + 16 * 1024 * 4 + 16 * 1024 * 2);// 256 f32
  float*  sRow = sRed + 256;                                    // 16 f32
  float*  sPar = sRow + 16;                                     // 4*16*16 f32

  const int qt = blockIdx.x, h = blockIdx.y, b = blockIdx.z;
  const int tid = threadIdx.x, lane = tid & 31, wave = tid >> 5;
  const int bbase = b << 10;
  const int q0 = qt << 4;
  const int half = lane >> 4;
  const int nidx = lane & 15;

  // Phase 1: scores tile (16 x 1024) via WMMA, 8 key-tiles per wave.
  for (int kt = wave * 8; kt < wave * 8 + 8; ++kt) {
    v8f acc = {};
#pragma unroll
    for (int kc = 0; kc < HDN; kc += 32) {
      v16bf a = load_a_frag(qb + (size_t)(bbase + q0) * DN + h * HDN + kc, DN,
                            lane);
      v16bf bf = load_b_frag_nk(kb + (size_t)(bbase + kt * 16) * DN + h * HDN +
                                kc, DN, lane);
      acc = wmma_bf16(a, bf, acc);
    }
#pragma unroll
    for (int r = 0; r < 8; ++r) {
      const int m = r + 8 * half;
      const int key = kt * 16 + nidx;
      sS[m * 1024 + key] = acc[r] * 0.125f + biasv[bbase + key];
    }
  }
  __syncthreads();

  // Phase 2: row-wise softmax in LDS (16 rows x 1024, 16 threads per row).
  {
    const int row = tid >> 4, seg = tid & 15;
    float* srow = sS + row * 1024;
    float lmax = -3.4e38f;
    for (int c = seg * 64; c < seg * 64 + 64; ++c) lmax = fmaxf(lmax, srow[c]);
    sRed[tid] = lmax;
    __syncthreads();
    float rmax = -3.4e38f;
#pragma unroll
    for (int i = 0; i < 16; ++i) rmax = fmaxf(rmax, sRed[row * 16 + i]);
    __syncthreads();
    float lsum = 0.0f;
    for (int c = seg * 64; c < seg * 64 + 64; ++c) {
      const float e = __expf(srow[c] - rmax);
      sP[row * 1024 + c] = (bf16_t)e;
      lsum += e;
    }
    sRed[tid] = lsum;
    __syncthreads();
    if (seg == 0) {
      float s = 0.0f;
#pragma unroll
      for (int i = 0; i < 16; ++i) s += sRed[row * 16 + i];
      sRow[row] = s;
    }
    __syncthreads();
  }

  // Phase 3: O = P @ V via WMMA; 4 n-tiles x 2 K-halves across 8 waves.
  const int ntile = wave & 3;
  const int khalf = wave >> 2;
  v8f acc = {};
  for (int kk0 = khalf * 512; kk0 < khalf * 512 + 512; kk0 += 32) {
    v16bf a  = load_a_frag(sP + kk0, 1024, lane);  // probs from LDS
    v16bf bf = load_b_frag_kn(vb + (size_t)(bbase + kk0) * DN + h * HDN +
                              ntile * 16, DN, lane);
    acc = wmma_bf16(a, bf, acc);
  }
  if (khalf == 1) {
#pragma unroll
    for (int r = 0; r < 8; ++r)
      sPar[(ntile * 16 + r + 8 * half) * 16 + nidx] = acc[r];
  }
  __syncthreads();
  if (khalf == 0) {
#pragma unroll
    for (int r = 0; r < 8; ++r) {
      const int m = r + 8 * half;
      const float v =
          (acc[r] + sPar[(ntile * 16 + m) * 16 + nidx]) / sRow[m];
      ob[(size_t)(bbase + q0 + m) * DN + h * HDN + ntile * 16 + nidx] =
          (bf16_t)v;
    }
  }
}

// ---------------- residual + LayerNorm -------------------------------------
__global__ __launch_bounds__(256) void k_res_ln(
    const float* __restrict__ xin, const float* __restrict__ add,
    const float* __restrict__ g, const float* __restrict__ bb,
    float* __restrict__ xoutf, bf16_t* __restrict__ xoutb) {
  __shared__ float red[256];
  const int t = blockIdx.x, tid = threadIdx.x;
  const size_t base = (size_t)t * DN;
  const float v0 = xin[base + tid] + add[base + tid];
  const float v1 = xin[base + tid + 256] + add[base + tid + 256];
  red[tid] = v0 + v1;
  __syncthreads();
  for (int s = 128; s > 0; s >>= 1) {
    if (tid < s) red[tid] += red[tid + s];
    __syncthreads();
  }
  const float mean = red[0] * (1.0f / 512.0f);
  __syncthreads();
  const float d0 = v0 - mean, d1 = v1 - mean;
  red[tid] = d0 * d0 + d1 * d1;
  __syncthreads();
  for (int s = 128; s > 0; s >>= 1) {
    if (tid < s) red[tid] += red[tid + s];
    __syncthreads();
  }
  const float rstd = rsqrtf(red[0] * (1.0f / 512.0f) + 1e-5f);
  const float o0 = d0 * rstd * g[tid] + bb[tid];
  const float o1 = d1 * rstd * g[tid + 256] + bb[tid + 256];
  xoutf[base + tid] = o0;
  xoutf[base + tid + 256] = o1;
  xoutb[base + tid] = (bf16_t)o0;
  xoutb[base + tid + 256] = (bf16_t)o1;
}

// ---------------- compact target rows --------------------------------------
__global__ void k_copy_tgt(const bf16_t* __restrict__ xb,
                           bf16_t* __restrict__ tgtb) {
  const int r = blockIdx.x;  // 0..8191
  const int src = ((r >> 9) << 10) + 512 + (r & 511);
  for (int d = threadIdx.x; d < DN; d += 256)
    tgtb[(size_t)r * DN + d] = xb[(size_t)src * DN + d];
}

// ---------------------------------------------------------------------------
extern "C" void kernel_launch(void* const* d_in, const int* in_sizes, int n_in,
                              void* d_out, int out_size, void* d_ws,
                              size_t ws_size, hipStream_t stream) {
  const int* ccat = (const int*)d_in[0];
  const float* ctime = (const float*)d_in[1];
  const float* cf0 = (const float*)d_in[2];
  const int* tcat = (const int*)d_in[3];
  const float* ttime = (const float*)d_in[4];
  const float* tf0 = (const float*)d_in[5];
  const float* cat_table = (const float*)d_in[6];
  const float* cont_table = (const float*)d_in[7];
  const float* pad_emb = (const float*)d_in[8];
  const float* Wq = (const float*)d_in[9];
  const float* bq = (const float*)d_in[10];
  const float* Wk = (const float*)d_in[11];
  const float* bk = (const float*)d_in[12];
  const float* Wv = (const float*)d_in[13];
  const float* bv = (const float*)d_in[14];
  const float* Wo = (const float*)d_in[15];
  const float* bo = (const float*)d_in[16];
  const float* ln1g = (const float*)d_in[17];
  const float* ln1b = (const float*)d_in[18];
  const float* ln2g = (const float*)d_in[19];
  const float* ln2b = (const float*)d_in[20];
  const float* W1 = (const float*)d_in[21];
  const float* b1 = (const float*)d_in[22];
  const float* W2 = (const float*)d_in[23];
  const float* b2 = (const float*)d_in[24];
  float* out = (float*)d_out;

  // workspace partition
  char* wp = (char*)d_ws;
  auto alloc = [&](size_t bytes) -> void* {
    void* p = wp;
    wp += (bytes + 255) & ~(size_t)255;
    return p;
  };
  float*  xf    = (float*)alloc((size_t)MTOK * DN * 4);
  bf16_t* xb    = (bf16_t*)alloc((size_t)MTOK * DN * 2);
  float*  biasv = (float*)alloc((size_t)MTOK * 4);
  bf16_t* qbuf  = (bf16_t*)alloc((size_t)MTOK * DN * 2);
  bf16_t* kbuf  = (bf16_t*)alloc((size_t)MTOK * DN * 2);
  bf16_t* vbuf  = (bf16_t*)alloc((size_t)MTOK * DN * 2);
  bf16_t* abuf  = (bf16_t*)alloc((size_t)MTOK * DN * 2);
  float*  tmpf  = (float*)alloc((size_t)MTOK * DN * 4);
  bf16_t* h1b   = (bf16_t*)alloc((size_t)MTOK * DFFN * 2);
  bf16_t* tgtb  = (bf16_t*)alloc((size_t)MTGT * DN * 2);
  bf16_t* wqb = (bf16_t*)alloc((size_t)LN_ * DN * DN * 2);
  bf16_t* wkb = (bf16_t*)alloc((size_t)LN_ * DN * DN * 2);
  bf16_t* wvb = (bf16_t*)alloc((size_t)LN_ * DN * DN * 2);
  bf16_t* wob = (bf16_t*)alloc((size_t)LN_ * DN * DN * 2);
  bf16_t* w1b = (bf16_t*)alloc((size_t)LN_ * DN * DFFN * 2);
  bf16_t* w2b = (bf16_t*)alloc((size_t)LN_ * DFFN * DN * 2);
  bf16_t* catb  = (bf16_t*)alloc((size_t)NCATN * 256 * 2);
  bf16_t* contb = (bf16_t*)alloc((size_t)NCONTN * 128 * 2);

  auto cvt = [&](const float* s, bf16_t* d, size_t n) {
    k_cvt<<<dim3((unsigned)((n + 255) / 256)), 256, 0, stream>>>(s, d, (int)n);
  };
  auto cvtT = [&](const float* s, bf16_t* d, int K, int N) {
    const int tot = K * N;
    k_cvt_t<<<dim3((unsigned)((tot + 255) / 256)), 256, 0, stream>>>(s, d, K,
                                                                     N);
  };
  auto gemm = [&](const bf16_t* A, int lda, const bf16_t* Wt, int ldw,
                  const float* bias, float* Cf, bf16_t* Cb, int ldc, int M,
                  int N, int K, bool relu) {
    dim3 grid((unsigned)((N + 511) / 512), (unsigned)(M / 16));
    if (relu)
      k_gemm<1><<<grid, 256, 0, stream>>>(A, lda, Wt, ldw, bias, Cf, Cb, ldc,
                                          M, N, K);
    else
      k_gemm<0><<<grid, 256, 0, stream>>>(A, lda, Wt, ldw, bias, Cf, Cb, ldc,
                                          M, N, K);
  };

  // 1) embeddings + pad bias
  k_embed<<<dim3(MTOK), 256, 0, stream>>>(ccat, ctime, cf0, tcat, ttime, tf0,
                                          cat_table, pad_emb, xf, xb, biasv);

  // 2) weight conversions: per-layer transpose to [N,K] bf16
  for (int l = 0; l < LN_; ++l) {
    const size_t wdd = (size_t)l * DN * DN;
    const size_t wdf = (size_t)l * DN * DFFN;
    cvtT(Wq + wdd, wqb + wdd, DN, DN);
    cvtT(Wk + wdd, wkb + wdd, DN, DN);
    cvtT(Wv + wdd, wvb + wdd, DN, DN);
    cvtT(Wo + wdd, wob + wdd, DN, DN);
    cvtT(W1 + wdf, w1b + wdf, DN, DFFN);   // -> [DFF, D]
    cvtT(W2 + wdf, w2b + wdf, DFFN, DN);   // -> [D, DFF]
  }
  cvt(cat_table, catb, (size_t)NCATN * 256);     // already [N,K]
  cvt(cont_table, contb, (size_t)NCONTN * 128);  // already [N,K]

  const size_t smemAttn =
      (size_t)16 * 1024 * 4 + 16 * 1024 * 2 + (256 + 16 + 4 * 256) * 4;

  // 3) transformer layers
  for (int l = 0; l < LN_; ++l) {
    const size_t wdd = (size_t)l * DN * DN;
    const size_t wdf = (size_t)l * DN * DFFN;
    gemm(xb, DN, wqb + wdd, DN, bq + l * DN, nullptr, qbuf, DN, MTOK, DN, DN,
         false);
    gemm(xb, DN, wkb + wdd, DN, bk + l * DN, nullptr, kbuf, DN, MTOK, DN, DN,
         false);
    gemm(xb, DN, wvb + wdd, DN, bv + l * DN, nullptr, vbuf, DN, MTOK, DN, DN,
         false);
    k_attn<<<dim3(SN / 16, HN, BN), 256, smemAttn, stream>>>(qbuf, kbuf, vbuf,
                                                             biasv, abuf);
    gemm(abuf, DN, wob + wdd, DN, bo + l * DN, tmpf, nullptr, DN, MTOK, DN, DN,
         false);
    k_res_ln<<<dim3(MTOK), 256, 0, stream>>>(xf, tmpf, ln1g + l * DN,
                                             ln1b + l * DN, xf, xb);
    gemm(xb, DN, w1b + wdf, DN, b1 + l * DFFN, nullptr, h1b, DFFN, MTOK, DFFN,
         DN, true);
    gemm(h1b, DFFN, w2b + wdf, DFFN, b2 + l * DN, tmpf, nullptr, DN, MTOK, DN,
         DFFN, false);
    k_res_ln<<<dim3(MTOK), 256, 0, stream>>>(xf, tmpf, ln2g + l * DN,
                                             ln2b + l * DN, xf, xb);
  }

  // 4) gather target rows, then the three output-head GEMMs
  k_copy_tgt<<<dim3(MTGT), 256, 0, stream>>>(xb, tgtb);
  // pred_cat = out[:, :256] @ cat_table.T  (N=1000, guarded tiles)
  gemm(tgtb, DN, catb, 256, nullptr, out, nullptr, NCATN, MTGT, NCATN, 256,
       false);
  // pred_time = out[:, 256:384] @ cont_table.T
  gemm(tgtb + 256, DN, contb, 128, nullptr, out + (size_t)MTGT * NCATN,
       nullptr, NCONTN, MTGT, NCONTN, 128, false);
  // pred_f0 = out[:, 384:512] @ cont_table.T
  gemm(tgtb + 384, DN, contb, 128, nullptr,
       out + (size_t)MTGT * NCATN + (size_t)MTGT * NCONTN, nullptr, NCONTN,
       MTGT, NCONTN, 128, false);
}
